// MSTRModel_22539988370098
// MI455X (gfx1250) — matrix operand order repaired and checked by reading, not compiled, not run on hardware
//
#include <hip/hip_runtime.h>
#include <math.h>

// ---------------------------------------------------------------------------
// MSTR model for MI455X (gfx1250), wave32 + WMMA.
//
// Roofline reasoning:
//  * QKV GEMMs (M=131072,N=256,K=256, 51.5 GFLOP) dominate compute.  HBM
//    traffic for the stage is ~530MB -> ~23us at 23.3 TB/s.  bf16 WMMA
//    (16x16x32, f32 accum) runs ~8x the f32 matrix rate (K=32 vs K=4 per
//    instruction), putting the GEMM at the memory roof instead of the f32
//    matrix ceiling.  f32 accumulation keeps ~0.3% worst-case relative error.
//  * mean(axis=1) of (summed @ Wp + bp) == mean(summed) @ Wp + bp, so the
//    17-GFLOP Wp GEMM over 131072 rows collapses to a [32,256]x[256,256] FC.
//  * Attention windows are 4x4 per head -> pure VALU, bandwidth bound.
//  * Wt staging into LDS is a pure bf16 copy -> CDNA5 async load-to-LDS
//    (ASYNCcnt path, global_load_async_to_lds_b128), no VGPR round-trip.
// ---------------------------------------------------------------------------

typedef unsigned int       u32;
typedef unsigned short     u16;
typedef __attribute__((ext_vector_type(16))) __bf16 v16bf;
typedef __attribute__((ext_vector_type(8)))  float  v8f;
typedef __attribute__((ext_vector_type(4)))  u32    v4u;
typedef __attribute__((ext_vector_type(4)))  int    v4i;

#if defined(__has_builtin)
#if __has_builtin(__builtin_amdgcn_global_load_async_to_lds_b128) && \
    __has_builtin(__builtin_amdgcn_s_wait_asynccnt)
#define USE_ASYNC_LDS 1
#endif
#endif
#ifndef USE_ASYNC_LDS
#define USE_ASYNC_LDS 0
#endif

#if USE_ASYNC_LDS
// builtin signature (from hipcc diagnostic): arg0 = AS1 (__device__) v4i*,
// arg1 = AS3 (__shared__) v4i*, arg2/arg3 = imm offset / cpol.
typedef __attribute__((address_space(1))) v4i gv4i;
typedef __attribute__((address_space(3))) v4i sv4i;
#endif

#define BATCH 32
#define TLEN  4096
#define DIM   256
#define MROWS ((size_t)BATCH * TLEN)          // 131072
#define HEADS 4
#define WSZ   4

// round-to-nearest-even fp32 -> bf16
static __device__ __forceinline__ u16 f2bf(float x) {
    u32 u = __builtin_bit_cast(u32, x);
    u32 r = u + 0x7FFFu + ((u >> 16) & 1u);
    return (u16)(r >> 16);
}

struct Frag128x2 { v4u a, b; };   // 32 bytes == sizeof(v16bf)

// two ds_load_b128 from LDS -> one 16xbf16 WMMA fragment
static __device__ __forceinline__ v16bf load_frag(const u16* p0, const u16* p1) {
    Frag128x2 r;
    r.a = *(const v4u*)p0;
    r.b = *(const v4u*)p1;
    return __builtin_bit_cast(v16bf, r);
}

static __device__ __forceinline__ float gelu_exact(float x) {
    return 0.5f * x * (1.0f + erff(x * 0.70710678118654752f));
}

// ---------------------------------------------------------------------------
// 0) Weight transpose + bf16 convert:  Wt[n*256+k] = bf16(W[k*256+n])
// ---------------------------------------------------------------------------
__global__ __launch_bounds__(256) void wconv_kernel(const float* __restrict__ W,
                                                    u16* __restrict__ Wt) {
    int n = blockIdx.x;        // 0..255  (output row = column of W)
    int k = threadIdx.x;       // 0..255
    Wt[n * DIM + k] = f2bf(W[k * DIM + n]);
}

// ---------------------------------------------------------------------------
// 1) Fused QKV GEMM.  grid = (M/64, 3), block = 256 (8 waves).
//    Each block: 64 rows x 256 cols of one of Q/K/V.
//    LDS k-chunk = 32, bf16, padded stride 40 (80B, keeps 16B alignment).
//    Wave w: M-tile = w&3 (16 rows), N-tiles = (w>>2)*8 .. +7.
// ---------------------------------------------------------------------------
#define GEMM_BM 64
#define GEMM_KC 32
#define LDS_STR 40

__global__ __launch_bounds__(256) void qkv_gemm_kernel(
        const float* __restrict__ x,    // [M,256] fp32
        const u16*   __restrict__ Wt3,  // 3 x [256n][256k] bf16 (transposed)
        const float* __restrict__ bq,
        const float* __restrict__ bk,
        const float* __restrict__ bv,
        float* __restrict__ outbase)    // q|k|v contiguous, each [M,256]
{
    __shared__ u16 lX[GEMM_BM * LDS_STR];   //  5.0 KB
    __shared__ u16 lW[DIM    * LDS_STR];    // 20.0 KB

    const int tid  = threadIdx.x;
    const int lane = tid & 31;
    const int wave = tid >> 5;
    const int mt   = wave & 3;           // M-tile within block
    const int ntb  = (wave >> 2) * 8;    // first N-tile of this wave

    const size_t m0  = (size_t)blockIdx.x * GEMM_BM;
    const int    mat = blockIdx.y;                       // 0=q 1=k 2=v
    const u16*   Wt  = Wt3 + (size_t)mat * DIM * DIM;
    const float* bias = (mat == 0) ? bq : ((mat == 1) ? bk : bv);
    float*       out  = outbase + (size_t)mat * MROWS * DIM;

    v8f acc[8] = {};   // 8 N-tiles x (16x16 f32 tile / 32 lanes = 8 VGPRs)

    const int l15   = lane & 15;
    const int asel  = (lane < 16) ? 0 : 8;    // A frag K-half select (elements)
    const int bsel  = (lane < 16) ? 0 : 8;    // B frag K-half select (elements)
    const int arow  = mt * 16 + l15;

    for (int k0 = 0; k0 < DIM; k0 += GEMM_KC) {
#if USE_ASYNC_LDS
        // --- stage Wt chunk via async copy to LDS (pure bf16 copy, no VGPRs)
        // 256 rows x 32 k = 1024 x b128 chunks, 4 per thread.
        #pragma unroll
        for (int r = 0; r < 4; ++r) {
            int i   = tid + r * 256;           // 0..1023
            int nn  = i >> 2, kp8 = i & 3;     // 4x (8 bf16 = 16B) per row
            __builtin_amdgcn_global_load_async_to_lds_b128(
                (gv4i*)(Wt + nn * DIM + k0 + kp8 * 8),
                (sv4i*)(lW + nn * LDS_STR + kp8 * 8),
                0, 0);
        }
#else
        // --- stage Wt chunk (256x32 bf16), 16 u32 / thread ---
        #pragma unroll
        for (int r = 0; r < 16; ++r) {
            int i  = tid + r * 256;            // 0..4095
            int nn = i >> 4, kp = i & 15;      // kp = bf16 pair index
            u32 wv = *(const u32*)(Wt + nn * DIM + k0 + kp * 2);
            *((u32*)(lW + nn * LDS_STR + kp * 2)) = wv;
        }
#endif
        // --- stage x chunk (64x32 fp32 -> bf16), 8 elements / thread ---
        #pragma unroll
        for (int r = 0; r < 8; ++r) {
            int i  = tid + r * 256;            // 0..2047
            int mm = i >> 5, kk = i & 31;
            lX[mm * LDS_STR + kk] = f2bf(x[(m0 + mm) * DIM + k0 + kk]);
        }
        // prefetch next x chunk into cache (global_prefetch_b8)
        if (k0 + GEMM_KC < DIM) {
            __builtin_prefetch(&x[(m0 + (tid >> 3)) * DIM + k0 + GEMM_KC], 0, 3);
        }
#if USE_ASYNC_LDS
        __builtin_amdgcn_s_wait_asynccnt(0);
#endif
        __syncthreads();

        // A fragment: lane<16 holds K{0..7,16..23}, lane>=16 K{8..15,24..31}
        v16bf afrag = load_frag(&lX[arow * LDS_STR + asel],
                                &lX[arow * LDS_STR + asel + 16]);
        #pragma unroll
        for (int j = 0; j < 8; ++j) {
            int col = (ntb + j) * 16 + l15;    // B column = lane%16
            // B fragment: lane<16 holds K{0..7,16..23}, lane>=16 K{8..15,24..31}
            v16bf bfrag = load_frag(&lW[col * LDS_STR + bsel],
                                    &lW[col * LDS_STR + bsel + 16]);
            acc[j] = __builtin_amdgcn_wmma_f32_16x16x32_bf16(
                         false, afrag, false, bfrag,
                         (short)0, acc[j], false, false);
        }
        __syncthreads();
    }

    // epilogue: C layout -> lanes 0-15: M=r, N=lane; lanes 16-31: M=8+r, N=lane-16
    const int rbase = (lane < 16) ? 0 : 8;
    #pragma unroll
    for (int j = 0; j < 8; ++j) {
        int col = (ntb + j) * 16 + l15;
        float bvv = bias[col];
        #pragma unroll
        for (int r = 0; r < 8; ++r) {
            size_t m = m0 + mt * 16 + rbase + r;
            out[m * DIM + col] = acc[j][r] + bvv;
        }
    }
}

// ---------------------------------------------------------------------------
// 2) Reflect-pad average pool (ksize 2 or 4) on q,k,v simultaneously.
//    One thread per (b, l, float4-chunk).
// ---------------------------------------------------------------------------
__global__ __launch_bounds__(256) void pool_kernel(
        const float* __restrict__ q, const float* __restrict__ k,
        const float* __restrict__ v,
        float* __restrict__ qp, float* __restrict__ kp, float* __restrict__ vp,
        int ksize, int L, int total)
{
    int idx = blockIdx.x * 256 + threadIdx.x;
    if (idx >= total) return;
    int f4   = idx & 63;
    int rest = idx >> 6;
    int l    = rest % L;
    int b    = rest / L;
    int p    = ksize >> 1;

    float qx=0,qy=0,qz=0,qw=0, kx=0,ky=0,kz=0,kw=0, vx=0,vy=0,vz=0,vw=0;
    for (int j = 0; j < ksize; ++j) {
        int pos = l * ksize + j - p;
        if (pos < 0)      pos = -pos;
        if (pos >= TLEN)  pos = 2 * TLEN - 2 - pos;
        size_t off = ((size_t)b * TLEN + pos) * DIM + f4 * 4;
        float4 a = *(const float4*)(q + off);
        float4 c = *(const float4*)(k + off);
        float4 d = *(const float4*)(v + off);
        qx+=a.x; qy+=a.y; qz+=a.z; qw+=a.w;
        kx+=c.x; ky+=c.y; kz+=c.z; kw+=c.w;
        vx+=d.x; vy+=d.y; vz+=d.z; vw+=d.w;
    }
    float inv = 1.0f / (float)ksize;
    size_t o = ((size_t)b * L + l) * DIM + f4 * 4;
    float4 r;
    r.x=qx*inv; r.y=qy*inv; r.z=qz*inv; r.w=qw*inv; *(float4*)(qp+o)=r;
    r.x=kx*inv; r.y=ky*inv; r.z=kz*inv; r.w=kw*inv; *(float4*)(kp+o)=r;
    r.x=vx*inv; r.y=vy*inv; r.z=vz*inv; r.w=vw*inv; *(float4*)(vp+o)=r;
}

// ---------------------------------------------------------------------------
// 3) Windowed multi-head attention (W=4, H=4, hd=64), zero-pad semantics:
//    padded K rows contribute score 0 (pre-softmax), padded V rows are zero.
//    One thread per (b, window, head, qpos); 8 windows per 128-thread block.
// ---------------------------------------------------------------------------
__global__ __launch_bounds__(128) void win_attn_kernel(
        const float* __restrict__ q, const float* __restrict__ k,
        const float* __restrict__ v, float* __restrict__ out,
        int Ts, int nw, int nWinTotal)
{
    int tid    = threadIdx.x;
    int wlocal = tid >> 4;
    int lane16 = tid & 15;
    int h = lane16 >> 2;
    int i = lane16 & 3;
    int windex = blockIdx.x * 8 + wlocal;
    if (windex >= nWinTotal) return;
    int b   = windex / nw;
    int win = windex - b * nw;
    int t   = win * WSZ + i;
    if (t >= Ts) return;

    size_t rowQ = ((size_t)b * Ts + t) * DIM + h * 64;
    const float4* qr = (const float4*)(q + rowQ);
    float4 qreg[16];
    #pragma unroll
    for (int d = 0; d < 16; ++d) qreg[d] = qr[d];

    float s[WSZ];
    #pragma unroll
    for (int j = 0; j < WSZ; ++j) {
        int tj = win * WSZ + j;
        float acc = 0.0f;
        if (tj < Ts) {
            const float4* kr = (const float4*)(k + ((size_t)b * Ts + tj) * DIM + h * 64);
            #pragma unroll
            for (int d = 0; d < 16; ++d) {
                float4 a = qreg[d], c = kr[d];
                acc += a.x*c.x + a.y*c.y + a.z*c.z + a.w*c.w;
            }
            acc *= 0.125f;                       // 1/sqrt(64)
        }
        s[j] = acc;                              // padded slot: exact 0
    }

    float mx = fmaxf(fmaxf(s[0], s[1]), fmaxf(s[2], s[3]));
    float e[WSZ]; float sum = 0.0f;
    #pragma unroll
    for (int j = 0; j < WSZ; ++j) { e[j] = expf(s[j] - mx); sum += e[j]; }
    float inv = 1.0f / sum;
    #pragma unroll
    for (int j = 0; j < WSZ; ++j) e[j] *= inv;

    float4* outr = (float4*)(out + rowQ);
    #pragma unroll
    for (int d = 0; d < 16; ++d) {
        float ox=0, oy=0, oz=0, ow=0;
        #pragma unroll
        for (int j = 0; j < WSZ; ++j) {
            int tj = win * WSZ + j;
            if (tj < Ts) {
                float4 c = *(const float4*)(v + ((size_t)b * Ts + tj) * DIM + h * 64 + d * 4);
                ox += e[j]*c.x; oy += e[j]*c.y; oz += e[j]*c.z; ow += e[j]*c.w;
            }
        }
        float4 o; o.x=ox; o.y=oy; o.z=oz; o.w=ow;
        outr[d] = o;
    }
}

// ---------------------------------------------------------------------------
// 4) Nearest-upsample index + exact GELU + time-mean, accumulated into h0.
//    mean_t gelu(f[ floor(t*Ls/4096) ][ft]) ; thread per (b, feature).
// ---------------------------------------------------------------------------
__global__ __launch_bounds__(256) void mean_gelu_kernel(
        const float* __restrict__ f, float* __restrict__ h0,
        int Ls, int accumulate)
{
    int idx = blockIdx.x * 256 + threadIdx.x;   // 0..8191
    int b  = idx >> 8;
    int ft = idx & 255;
    const float* base = f + (size_t)b * Ls * DIM + ft;
    float acc = 0.0f;
    for (int t = 0; t < TLEN; ++t) {
        int l = (int)(((long long)t * Ls) >> 12);    // t*Ls/4096
        if (l > Ls - 1) l = Ls - 1;
        acc += gelu_exact(base[(size_t)l * DIM]);
    }
    acc *= (1.0f / (float)TLEN);
    if (accumulate) h0[idx] += acc;
    else            h0[idx]  = acc;
}

// ---------------------------------------------------------------------------
// 5) Small FC: out[r][n] = act(sum_k in[r][k]*W[k][n] + b[n]), rows=32.
// ---------------------------------------------------------------------------
__global__ __launch_bounds__(256) void fc_kernel(
        const float* __restrict__ in, const float* __restrict__ W,
        const float* __restrict__ b, float* __restrict__ out,
        int K, int N, int relu_act)
{
    int idx = blockIdx.x * 256 + threadIdx.x;
    if (idx >= BATCH * N) return;
    int n   = idx % N;
    int row = idx / N;
    float s = b[n];
    for (int k = 0; k < K; ++k) s += in[row * K + k] * W[k * N + n];
    if (relu_act) s = fmaxf(s, 0.0f);
    out[row * N + n] = s;
}

// ---------------------------------------------------------------------------
// launcher
// ---------------------------------------------------------------------------
extern "C" void kernel_launch(void* const* d_in, const int* in_sizes, int n_in,
                              void* d_out, int out_size, void* d_ws, size_t ws_size,
                              hipStream_t stream) {
    const float* x   = (const float*)d_in[0];
    const float* Wq  = (const float*)d_in[1];
    const float* bq  = (const float*)d_in[2];
    const float* Wk  = (const float*)d_in[3];
    const float* bk  = (const float*)d_in[4];
    const float* Wv  = (const float*)d_in[5];
    const float* bv  = (const float*)d_in[6];
    const float* Wp  = (const float*)d_in[7];
    const float* bp  = (const float*)d_in[8];
    const float* Wc1 = (const float*)d_in[9];
    const float* bc1 = (const float*)d_in[10];
    const float* Wc2 = (const float*)d_in[11];
    const float* bc2 = (const float*)d_in[12];
    float* logits = (float*)d_out;

    // ---- workspace layout ----
    const size_t MT   = MROWS * DIM;               // 33,554,432 floats / tensor
    const size_t PCAP = (size_t)BATCH * 2049 * DIM; // pooled capacity (largest)
    float* qbuf  = (float*)d_ws;                   // q|k|v contiguous (3*MT)
    float* kbuf  = qbuf + MT;
    float* vbuf  = kbuf + MT;
    u16*   wt    = (u16*)(vbuf + MT);              // 3 * 256*256 bf16
    float* poolq = (float*)((char*)wt + (size_t)3 * DIM * DIM * sizeof(u16));
    float* poolk = poolq + PCAP;
    float* poolv = poolk + PCAP;
    float* attnb = poolv + PCAP;                   // MT floats (reused per scale)
    float* h0    = attnb + MT;                     // [32,256]
    float* h1    = h0 + BATCH * DIM;
    float* h2    = h1 + BATCH * DIM;

    // 0) weights -> bf16 transposed
    wconv_kernel<<<DIM, DIM, 0, stream>>>(Wq, wt + 0 * DIM * DIM);
    wconv_kernel<<<DIM, DIM, 0, stream>>>(Wk, wt + 1 * DIM * DIM);
    wconv_kernel<<<DIM, DIM, 0, stream>>>(Wv, wt + 2 * DIM * DIM);

    // 1) fused QKV GEMM (bf16 WMMA, f32 accum)
    dim3 ggrid((unsigned)(MROWS / GEMM_BM), 3);
    qkv_gemm_kernel<<<ggrid, 256, 0, stream>>>(x, wt, bq, bk, bv, qbuf);

    // ---- scale 1 (ksize=1, Ts=4096) ----
    {
        int Ts = TLEN, nw = Ts / WSZ, nWin = BATCH * nw;
        win_attn_kernel<<<(nWin + 7) / 8, 128, 0, stream>>>(qbuf, kbuf, vbuf, attnb, Ts, nw, nWin);
        mean_gelu_kernel<<<(BATCH * DIM) / 256, 256, 0, stream>>>(attnb, h0, Ts, 0);
    }
    // ---- scale 2 (ksize=2, L=2049) ----
    {
        int L = 2049, total = BATCH * L * (DIM / 4);
        pool_kernel<<<(total + 255) / 256, 256, 0, stream>>>(qbuf, kbuf, vbuf,
                                                             poolq, poolk, poolv, 2, L, total);
        int nw = (L + WSZ - 1) / WSZ, nWin = BATCH * nw;
        win_attn_kernel<<<(nWin + 7) / 8, 128, 0, stream>>>(poolq, poolk, poolv, attnb, L, nw, nWin);
        mean_gelu_kernel<<<(BATCH * DIM) / 256, 256, 0, stream>>>(attnb, h0, L, 1);
    }
    // ---- scale 3 (ksize=4, L=1025) ----
    {
        int L = 1025, total = BATCH * L * (DIM / 4);
        pool_kernel<<<(total + 255) / 256, 256, 0, stream>>>(qbuf, kbuf, vbuf,
                                                             poolq, poolk, poolv, 4, L, total);
        int nw = (L + WSZ - 1) / WSZ, nWin = BATCH * nw;
        win_attn_kernel<<<(nWin + 7) / 8, 128, 0, stream>>>(poolq, poolk, poolv, attnb, L, nw, nWin);
        mean_gelu_kernel<<<(BATCH * DIM) / 256, 256, 0, stream>>>(attnb, h0, L, 1);
    }

    // classifier head (mean-linearity: Wp applied AFTER time mean)
    fc_kernel<<<(BATCH * DIM + 255) / 256, 256, 0, stream>>>(h0, Wp,  bp,  h1, DIM, DIM, 0);
    fc_kernel<<<(BATCH * DIM + 255) / 256, 256, 0, stream>>>(h1, Wc1, bc1, h2, DIM, DIM, 1);
    fc_kernel<<<1, 256, 0, stream>>>(h2, Wc2, bc2, logits, DIM, 7, 0);
}
